// ALiBi_85676007621286
// MI455X (gfx1250) — compile-verified, matrix operand order checked by reading, not run on hardware
//
#include <hip/hip_runtime.h>
#include <hip/hip_bf16.h>
#include <math.h>

// ---------------------------------------------------------------------------
// ALiBi causal attention layer for MI455X (gfx1250, wave32, WMMA bf16)
// B=2, T=2048, C=1024, H=16, D=64
// ---------------------------------------------------------------------------

typedef __attribute__((ext_vector_type(16))) __bf16        v16bf;
typedef __attribute__((ext_vector_type(8)))  float         v8f;
typedef __attribute__((ext_vector_type(4)))  unsigned int  v4u;

#define BB 2
#define TT 2048
#define CC 1024
#define HH 16
#define DD 64

__device__ __forceinline__ unsigned short f2bf(float f) {
  // round-to-nearest-even fp32 -> bf16
  unsigned u = __float_as_uint(f);
  unsigned r = u + 0x7FFFu + ((u >> 16) & 1u);
  return (unsigned short)(r >> 16);
}

union FragAB { v16bf f; v4u q[2]; };

__device__ __forceinline__ v8f wmma_bf16(const FragAB& a, const FragAB& b, v8f c) {
  return __builtin_amdgcn_wmma_f32_16x16x32_bf16(false, a.f, false, b.f,
                                                 (short)0, c, false, false);
}

// 16x32 bf16 fragment in the CDNA5 16-bit A-matrix lane layout:
//   lanes 0-15 : row r0+lane,    K = {k0+0..7,  k0+16..23}
//   lanes 16-31: row r0+lane-16, K = {k0+8..15, k0+24..31}
// (B-fragments of a transposed operand use the identical pattern with
//  "row" = output column, which is why weights are stored transposed.)
__device__ __forceinline__ FragAB load_frag(const __bf16* p, int r0, int k0, int ld) {
  const int lane = threadIdx.x & 31;
  const int r  = r0 + (lane & 15);
  const int ks = k0 + ((lane >> 4) << 3);
  FragAB fr;
  fr.q[0] = *reinterpret_cast<const v4u*>(p + (size_t)r * ld + ks);
  fr.q[1] = *reinterpret_cast<const v4u*>(p + (size_t)r * ld + ks + 16);
  return fr;
}

// ---------------------------------------------------------------------------
// fp32 -> bf16 conversion kernels
// ---------------------------------------------------------------------------
__global__ void cvt_bf16_kernel(const float* __restrict__ in,
                                unsigned short* __restrict__ out, int n) {
  int i = blockIdx.x * blockDim.x + threadIdx.x;
  if (i < n) out[i] = f2bf(in[i]);
}

// W:[K,N] fp32 -> Wt:[N,K] bf16 (transpose so B-fragments are contiguous)
__global__ void cvt_wt_kernel(const float* __restrict__ W,
                              unsigned short* __restrict__ Wt, int K, int N) {
  int idx = blockIdx.x * blockDim.x + threadIdx.x;
  if (idx < K * N) {
    int k = idx / N, n = idx % N;
    Wt[(size_t)n * K + k] = f2bf(W[idx]);
  }
}

// ---------------------------------------------------------------------------
// WMMA GEMM: OUT[M,N] = A[M,K] @ Bt[N,K]^T + bias
// 128x64 tile per 128-thread workgroup; each wave owns 32 rows x 64 cols
// (2 A-tiles x 4 B-tiles = 8 WMMAs/k-step, B-frag loads amortized over 2).
// Register double-buffering: k+32 fragments are in flight during k WMMAs.
// MODE 0: fp32 row-major    MODE 1: bf16 [B,H,T,D]    MODE 2: bf16 [B,H,D,T]
// ---------------------------------------------------------------------------
template <int MODE>
__global__ __launch_bounds__(128) void wmma_gemm_kernel(
    const __bf16* __restrict__ A, const __bf16* __restrict__ Bt,
    const float* __restrict__ bias, void* __restrict__ outp,
    int M, int N, int K) {
  const int lane = threadIdx.x & 31;
  const int wave = threadIdx.x >> 5;
  const int mb = blockIdx.y * 128 + wave * 32;
  const int nb = blockIdx.x * 64;

  v8f acc[2][4] = {};

  FragAB a0 = load_frag(A, mb, 0, K);
  FragAB a1 = load_frag(A, mb + 16, 0, K);
  FragAB b[4];
#pragma unroll
  for (int j = 0; j < 4; ++j) b[j] = load_frag(Bt, nb + 16 * j, 0, K);

  for (int kk = 0; kk < K; kk += 32) {
    const int kn = kk + 32;
    FragAB na0, na1, nb4[4];
    if (kn < K) {  // prefetch next k-step while current WMMAs run
      na0 = load_frag(A, mb, kn, K);
      na1 = load_frag(A, mb + 16, kn, K);
#pragma unroll
      for (int j = 0; j < 4; ++j) nb4[j] = load_frag(Bt, nb + 16 * j, kn, K);
    }
#pragma unroll
    for (int j = 0; j < 4; ++j) {
      acc[0][j] = wmma_bf16(a0, b[j], acc[0][j]);
      acc[1][j] = wmma_bf16(a1, b[j], acc[1][j]);
    }
    if (kn < K) {
      a0 = na0; a1 = na1;
#pragma unroll
      for (int j = 0; j < 4; ++j) b[j] = nb4[j];
    }
  }

  // C/D layout: VGPR i -> row (i or i+8), col = lane&15 per 16-col tile
  const int half = lane >> 4;
  const int nc = lane & 15;
#pragma unroll
  for (int r = 0; r < 2; ++r) {
#pragma unroll
    for (int j = 0; j < 4; ++j) {
      const int n = nb + 16 * j + nc;
      const float bv = bias[n];
#pragma unroll
      for (int i = 0; i < 8; ++i) {
        const int m = mb + 16 * r + i + half * 8;
        const float v = acc[r][j][i] + bv;
        if (MODE == 0) {
          ((float*)outp)[(size_t)m * N + n] = v;
        } else if (MODE == 1) {
          const int bi = m / TT, t = m % TT, h = n >> 6, d = n & 63;
          ((unsigned short*)outp)[(((size_t)bi * HH + h) * TT + t) * DD + d] = f2bf(v);
        } else {
          const int bi = m / TT, t = m % TT, h = n >> 6, d = n & 63;
          ((unsigned short*)outp)[(((size_t)bi * HH + h) * DD + d) * TT + t] = f2bf(v);
        }
      }
    }
  }
}

// ---------------------------------------------------------------------------
// Flash attention with ALiBi + causal mask. One wave per 16-query tile.
// Q,K: bf16 [B,H,T,D]; Vt: bf16 [B,H,D,T]; O: bf16 [B*T, C]
// K-fragments for the next key block and V-fragments for the current one are
// issued before the softmax VALU work so memory latency hides behind it.
// ---------------------------------------------------------------------------
__global__ __launch_bounds__(32) void flash_attn_kernel(
    const __bf16* __restrict__ Qh, const __bf16* __restrict__ Kh,
    const __bf16* __restrict__ Vth, unsigned short* __restrict__ O) {
  __shared__ __align__(16) unsigned short pbuf[16 * 32];

  const int lane = threadIdx.x & 31;
  const int q0 = blockIdx.x * 16;
  const int h = blockIdx.y;
  const int b = blockIdx.z;
  const int bh = b * HH + h;
  const __bf16* Qb = Qh + (size_t)bh * TT * DD;
  const __bf16* Kb = Kh + (size_t)bh * TT * DD;
  const __bf16* Vb = Vth + (size_t)bh * DD * TT;

  // H=16 (power of 2): slope[h] = 2^(-0.5*(h+1))
  const float slope = exp2f(-0.5f * (float)(h + 1));
  const float scale = 0.125f;  // 1/sqrt(D)
  const float L2E = 1.4426950408889634f;

  // Q tile fragments (D=64 -> two k-steps of 32), loaded once
  const FragAB qf0 = load_frag(Qb, q0, 0, DD);
  const FragAB qf1 = load_frag(Qb, q0, 32, DD);

  float mrow[8], lrow[8];
#pragma unroll
  for (int i = 0; i < 8; ++i) { mrow[i] = -1e30f; lrow[i] = 0.f; }
  v8f acc[4] = {};  // O accumulator 16x64 fp32 (4 d-tiles)

  const int half = lane >> 4;
  const int nc = lane & 15;

  FragAB kf[4];  // current key block's K fragments: [ntile][kstep]
  kf[0] = load_frag(Kb, 0, 0, DD);
  kf[1] = load_frag(Kb, 0, 32, DD);
  kf[2] = load_frag(Kb, 16, 0, DD);
  kf[3] = load_frag(Kb, 16, 32, DD);

  for (int kb = 0; kb <= q0 + 15; kb += 32) {
    // --- S = Q @ K^T for 32 keys (2 n-tiles x 2 k-steps) ---
    v8f s0 = {}, s1 = {};
    s0 = wmma_bf16(qf0, kf[0], s0);
    s0 = wmma_bf16(qf1, kf[1], s0);
    s1 = wmma_bf16(qf0, kf[2], s1);
    s1 = wmma_bf16(qf1, kf[3], s1);

    // --- prefetch next K block + this block's V while softmax runs ---
    const bool more = (kb + 32) <= (q0 + 15);
    FragAB nkf[4];
    if (more) {
      nkf[0] = load_frag(Kb, kb + 32, 0, DD);
      nkf[1] = load_frag(Kb, kb + 32, 32, DD);
      nkf[2] = load_frag(Kb, kb + 48, 0, DD);
      nkf[3] = load_frag(Kb, kb + 48, 32, DD);
    }
    FragAB vf[4];
#pragma unroll
    for (int j = 0; j < 4; ++j) vf[j] = load_frag(Vb, 16 * j, kb, TT);

    // --- scale + ALiBi + causal mask + online softmax (fp32) ---
    const int col0 = kb + nc, col1 = col0 + 16;
    float p0[8], p1[8], sf[8];
#pragma unroll
    for (int i = 0; i < 8; ++i) {
      const int row = q0 + i + half * 8;
      float f0 = (col0 <= row) ? (s0[i] * scale + slope * (float)(col0 - row)) : -1e30f;
      float f1 = (col1 <= row) ? (s1[i] * scale + slope * (float)(col1 - row)) : -1e30f;
      // row max across the 16-lane half (rows live in lane&15 columns)
      float t = fmaxf(f0, f1);
#pragma unroll
      for (int md = 1; md < 16; md <<= 1) t = fmaxf(t, __shfl_xor(t, md));
      const float mnew = fmaxf(mrow[i], t);
      sf[i] = exp2f((mrow[i] - mnew) * L2E);
      p0[i] = exp2f((f0 - mnew) * L2E);
      p1[i] = exp2f((f1 - mnew) * L2E);
      float rs = p0[i] + p1[i];
#pragma unroll
      for (int md = 1; md < 16; md <<= 1) rs += __shfl_xor(rs, md);
      lrow[i] = lrow[i] * sf[i] + rs;
      mrow[i] = mnew;
    }
#pragma unroll
    for (int j = 0; j < 4; ++j)
#pragma unroll
      for (int i = 0; i < 8; ++i) acc[j][i] *= sf[i];

    // --- re-layout P (C-fragment) -> A-fragment via LDS ---
    __syncthreads();
#pragma unroll
    for (int i = 0; i < 8; ++i) {
      const int rl = i + half * 8;
      pbuf[rl * 32 + nc] = f2bf(p0[i]);
      pbuf[rl * 32 + 16 + nc] = f2bf(p1[i]);
    }
    __syncthreads();
    FragAB pf;
    {
      const int ks = half << 3;
      pf.q[0] = *reinterpret_cast<const v4u*>(&pbuf[nc * 32 + ks]);
      pf.q[1] = *reinterpret_cast<const v4u*>(&pbuf[nc * 32 + ks + 16]);
    }

    // --- O += P @ V  (Vt gives contiguous keys per d-column) ---
#pragma unroll
    for (int j = 0; j < 4; ++j) acc[j] = wmma_bf16(pf, vf[j], acc[j]);

    if (more) {
#pragma unroll
      for (int j = 0; j < 4; ++j) kf[j] = nkf[j];
    }
  }

  // --- normalize and store bf16 [B*T, C] ---
#pragma unroll
  for (int j = 0; j < 4; ++j) {
#pragma unroll
    for (int i = 0; i < 8; ++i) {
      const int row = q0 + i + half * 8;
      const int d = 16 * j + nc;
      const float v = acc[j][i] / lrow[i];
      O[((size_t)(b * TT + row)) * CC + h * DD + d] = f2bf(v);
    }
  }
}

// ---------------------------------------------------------------------------
// Host launch
// ---------------------------------------------------------------------------
extern "C" void kernel_launch(void* const* d_in, const int* in_sizes, int n_in,
                              void* d_out, int out_size, void* d_ws, size_t ws_size,
                              hipStream_t stream) {
  const float* x  = (const float*)d_in[0];
  const float* Wq = (const float*)d_in[1];
  const float* bq = (const float*)d_in[2];
  const float* Wk = (const float*)d_in[3];
  const float* bk = (const float*)d_in[4];
  const float* Wv = (const float*)d_in[5];
  const float* bv = (const float*)d_in[6];
  const float* Wo = (const float*)d_in[7];
  const float* bo = (const float*)d_in[8];
  float* out = (float*)d_out;

  const int M = BB * TT;                 // 4096
  const size_t NE_X = (size_t)M * CC;    // 4M elements
  const size_t NE_W = (size_t)CC * CC;   // 1M elements

  char* ws = (char*)d_ws;
  unsigned short* Xb  = (unsigned short*)ws; ws += NE_X * 2;
  unsigned short* Wqt = (unsigned short*)ws; ws += NE_W * 2;
  unsigned short* Wkt = (unsigned short*)ws; ws += NE_W * 2;
  unsigned short* Wvt = (unsigned short*)ws; ws += NE_W * 2;
  unsigned short* Wot = (unsigned short*)ws; ws += NE_W * 2;
  unsigned short* Qh  = (unsigned short*)ws; ws += NE_X * 2;  // [B,H,T,D]
  unsigned short* Kh  = (unsigned short*)ws; ws += NE_X * 2;  // [B,H,T,D]
  unsigned short* Vth = (unsigned short*)ws; ws += NE_X * 2;  // [B,H,D,T]
  unsigned short* Oat = (unsigned short*)ws; ws += NE_X * 2;  // [B*T, C]

  cvt_bf16_kernel<<<dim3((unsigned)((NE_X + 255) / 256)), 256, 0, stream>>>(x, Xb, (int)NE_X);
  cvt_wt_kernel<<<dim3((unsigned)((NE_W + 255) / 256)), 256, 0, stream>>>(Wq, Wqt, CC, CC);
  cvt_wt_kernel<<<dim3((unsigned)((NE_W + 255) / 256)), 256, 0, stream>>>(Wk, Wkt, CC, CC);
  cvt_wt_kernel<<<dim3((unsigned)((NE_W + 255) / 256)), 256, 0, stream>>>(Wv, Wvt, CC, CC);
  cvt_wt_kernel<<<dim3((unsigned)((NE_W + 255) / 256)), 256, 0, stream>>>(Wo, Wot, CC, CC);

  dim3 gg(CC / 64, M / 128);  // (16, 32)
  wmma_gemm_kernel<1><<<gg, 128, 0, stream>>>((const __bf16*)Xb, (const __bf16*)Wqt, bq, Qh,  M, CC, CC);
  wmma_gemm_kernel<1><<<gg, 128, 0, stream>>>((const __bf16*)Xb, (const __bf16*)Wkt, bk, Kh,  M, CC, CC);
  wmma_gemm_kernel<2><<<gg, 128, 0, stream>>>((const __bf16*)Xb, (const __bf16*)Wvt, bv, Vth, M, CC, CC);

  dim3 ga(TT / 16, HH, BB);
  flash_attn_kernel<<<ga, 32, 0, stream>>>(
      (const __bf16*)Qh, (const __bf16*)Kh, (const __bf16*)Vth, Oat);

  wmma_gemm_kernel<0><<<gg, 128, 0, stream>>>((const __bf16*)Oat, (const __bf16*)Wot, bo, out, M, CC, CC);
}
